// GraphUNetWithSAGE_77446850281588
// MI455X (gfx1250) — matrix operand order repaired and checked by reading, not compiled, not run on hardware
//
#include <hip/hip_runtime.h>

// ---------------------------------------------------------------------------
// Graph U-Net (SAGE) for gfx1250 / MI455X.
//   - segment-mean aggregation via global f32 atomics (VMEM)
//   - all dense transforms via v_wmma_f32_16x16x32_bf16 (f32 accumulate),
//     LDS double-buffered; weights pre-converted to bf16 once per launch
//   - TopK pooling via stable O(n^2) LDS-tiled rank (matches jax.lax.top_k)
// ---------------------------------------------------------------------------

typedef __attribute__((ext_vector_type(16))) __bf16 v16bf;
typedef __attribute__((ext_vector_type(8)))  float  v8f;

union FragU { v16bf v; unsigned u[8]; };

#define K_N0 50000
#define K_E  800000
#define K_H  256
#define K_NS1 25000
#define K_NS2 12500
#define K_NS3 6250

__device__ __forceinline__ unsigned rne_bits(float f) {
    unsigned u = __float_as_uint(f);
    return u + 0x7fffu + ((u >> 16) & 1u);          // rounded bits, bf16 in [31:16]
}
__device__ __forceinline__ unsigned pack2bf(float a, float b) {
    unsigned ra = rne_bits(a), rb = rne_bits(b);
#if __has_builtin(__builtin_amdgcn_perm)
    // result = { rb[31:16], ra[31:16] } : one v_perm_b32
    return __builtin_amdgcn_perm(rb, ra, 0x07060302u);
#else
    return (ra >> 16) | (rb & 0xffff0000u);
#endif
}

// ---------------------------------------------------------------------------
// Fused dual-operand WMMA GEMM (weights already bf16):
//   C[M,256] = act( (A1 * diag(rs1)) @ W1^T + A2 @ W2^T + bias [+ C_prev] )
// W element (n,k) at W[n*ldw + k] (bf16).  Block tile 64x128, 8 waves of
// 32x32, double-buffered LDS, one barrier per 32-wide K step.
// ---------------------------------------------------------------------------
__global__ __launch_bounds__(256) void wmma_gemm_kernel(
    const float* __restrict__ A1, const float* __restrict__ rs1,
    const unsigned short* __restrict__ W1, int K1, int ldw1,
    const float* __restrict__ A2,
    const unsigned short* __restrict__ W2, int K2, int ldw2,
    const float* __restrict__ bias, float* __restrict__ C, int M,
    int do_relu, int addC)
{
    __shared__ unsigned sA[2][64 * 17];    // 64 rows x 32 bf16, stride 17 dw
    __shared__ unsigned sB[2][128 * 17];   // 128 cols x 32 bf16

    const int tid  = threadIdx.x;
    const int lane = tid & 31;
    const int wave = tid >> 5;
    const int wm = wave & 1;            // 2 wave-rows
    const int wn = wave >> 1;           // 4 wave-cols
    const int rowBase = blockIdx.x * 64;
    const int colBase = blockIdx.y * 128;

    v8f acc[2][2];
    #pragma unroll
    for (int i = 0; i < 2; ++i)
        #pragma unroll
        for (int j = 0; j < 2; ++j)
            #pragma unroll
            for (int v = 0; v < 8; ++v) acc[i][j][v] = 0.0f;

    const int nloc = lane & 15;
    const int rOff = (lane >> 4) * 8;
    if (addC) {
        #pragma unroll
        for (int i = 0; i < 2; ++i)
            #pragma unroll
            for (int j = 0; j < 2; ++j) {
                int col = colBase + wn * 32 + j * 16 + nloc;
                #pragma unroll
                for (int v = 0; v < 8; ++v) {
                    int row = rowBase + wm * 32 + i * 16 + rOff + v;
                    if (row < M) acc[i][j][v] = C[(long long)row * 256 + col];
                }
            }
    }

    // ---- staging geometry (per thread, constant across tiles) ----
    const int aRow = tid >> 2;              // 0..63
    const int aKg  = (tid & 3) * 8;         // 0,8,16,24
    const int aGr  = rowBase + aRow;
    const int aGrc = (aGr < M) ? aGr : (M - 1);
    const float aOk = (aGr < M) ? 1.0f : 0.0f;
    const int bN   = tid >> 1;              // 0..127
    const int bKg  = (tid & 1) * 16;        // 0,16
    const int bGn  = colBase + bN;

    const int t1 = K1 >> 5;
    const int t2 = A2 ? (K2 >> 5) : 0;
    const int total = t1 + t2;

    float4 ra0, ra1; float rsc;
    uint4 rbu0, rbu1;

    auto issueLoads = [&](int t) {
        const float *A, *rs; const unsigned short* W; int K, ldw, kt;
        if (t < t1) { A = A1; W = W1; rs = rs1; K = K1; ldw = ldw1; kt = t << 5; }
        else        { A = A2; W = W2; rs = 0;   K = K2; ldw = ldw2; kt = (t - t1) << 5; }
        float s = aOk;
        if (rs) s *= rs[aGrc];              // uniform null-check, in-bounds load
        const float4* ap = (const float4*)(A + (long long)aGrc * K + kt + aKg);
        ra0 = ap[0]; ra1 = ap[1]; rsc = s;
        const uint4* wp = (const uint4*)(W + (long long)bGn * ldw + kt + bKg);
        rbu0 = wp[0]; rbu1 = wp[1];         // 16 bf16, no conversion needed
    };
    auto stage = [&](int buf) {
        unsigned* dA = sA[buf] + aRow * 17 + (aKg >> 1);
        dA[0] = pack2bf(ra0.x * rsc, ra0.y * rsc);
        dA[1] = pack2bf(ra0.z * rsc, ra0.w * rsc);
        dA[2] = pack2bf(ra1.x * rsc, ra1.y * rsc);
        dA[3] = pack2bf(ra1.z * rsc, ra1.w * rsc);
        unsigned* dB = sB[buf] + bN * 17 + (bKg >> 1);
        dB[0] = rbu0.x; dB[1] = rbu0.y; dB[2] = rbu0.z; dB[3] = rbu0.w;
        dB[4] = rbu1.x; dB[5] = rbu1.y; dB[6] = rbu1.z; dB[7] = rbu1.w;
    };

    const int m   = lane & 15;
    const int kbA = (lane >> 4) * 8;    // A half-lane K base {0,8}
    const int kbB = (lane >> 4) * 16;   // B half-lane K base {0,8} (dword kbB>>2)

    issueLoads(0);
    stage(0);
    __syncthreads();

    for (int t = 0; t < total; ++t) {
        const int cur = t & 1;
        const bool more = (t + 1) < total;
        if (more) issueLoads(t + 1);

        FragU a[2], b[2];
        #pragma unroll
        for (int i = 0; i < 2; ++i) {
            const unsigned* uA = sA[cur] + (wm * 32 + i * 16 + m) * 17;
            #pragma unroll
            for (int v = 0; v < 4; ++v) a[i].u[v]     = uA[(kbA >> 1) + v];
            #pragma unroll
            for (int v = 0; v < 4; ++v) a[i].u[4 + v] = uA[8 + (kbA >> 1) + v];
        }
        #pragma unroll
        for (int j = 0; j < 2; ++j) {
            const unsigned* uB = sB[cur] + (wn * 32 + j * 16 + m) * 17;
            #pragma unroll
            for (int v = 0; v < 8; ++v) b[j].u[v] = uB[(kbB >> 2) + (v & 3) + ((v >> 2) << 3)];
        }

        #pragma unroll
        for (int i = 0; i < 2; ++i)
            #pragma unroll
            for (int j = 0; j < 2; ++j)
                acc[i][j] = __builtin_amdgcn_wmma_f32_16x16x32_bf16(
                    false, a[i].v, false, b[j].v,
                    (short)0, acc[i][j], false, false);

        if (more) stage((t + 1) & 1);
        __syncthreads();
    }

    #pragma unroll
    for (int i = 0; i < 2; ++i)
        #pragma unroll
        for (int j = 0; j < 2; ++j) {
            int col = colBase + wn * 32 + j * 16 + nloc;
            float bv = bias ? bias[col] : 0.0f;
            #pragma unroll
            for (int v = 0; v < 8; ++v) {
                int row = rowBase + wm * 32 + i * 16 + rOff + v;
                if (row < M) {
                    float val = acc[i][j][v] + bv;
                    if (do_relu) val = fmaxf(val, 0.0f);
                    C[(long long)row * 256 + col] = val;
                }
            }
        }
}

// --------------- one-shot f32 -> bf16 weight conversion --------------------
__global__ void cvt_bf16_kernel(const float* __restrict__ in,
                                unsigned* __restrict__ out, int nPairs)
{
    int i = blockIdx.x * blockDim.x + threadIdx.x;
    if (i < nPairs) out[i] = pack2bf(in[2 * i], in[2 * i + 1]);
}

// ----------------- edge aggregation (masked segment sum) -------------------
__global__ void agg_feat_kernel(const float* __restrict__ x,
                                const int* __restrict__ src,
                                const int* __restrict__ dst,
                                const float* __restrict__ mask,
                                float* __restrict__ acc, int E, int F)
{
    long long t = (long long)blockIdx.x * blockDim.x + threadIdx.x;
    int per = F >> 2;
    long long e = t / per;
    if (e >= E) return;
    int f = (int)(t - e * per) * 4;
    float mval = mask ? mask[e] : 1.0f;
    if (mval == 0.0f) return;
    int s = src[e], d = dst[e];
    const float4 xv = *(const float4*)(x + (long long)s * F + f);
    float* ap = acc + (long long)d * F + f;
    atomicAdd(ap + 0, xv.x * mval);
    atomicAdd(ap + 1, xv.y * mval);
    atomicAdd(ap + 2, xv.z * mval);
    atomicAdd(ap + 3, xv.w * mval);
}

__global__ void agg_cnt_kernel(const int* __restrict__ dst,
                               const float* __restrict__ mask,
                               float* __restrict__ cnt, int E)
{
    int e = blockIdx.x * blockDim.x + threadIdx.x;
    if (e >= E) return;
    float mval = mask ? mask[e] : 1.0f;
    if (mval != 0.0f) atomicAdd(&cnt[dst[e]], mval);
}

__global__ void rcnt_kernel(const float* __restrict__ cnt,
                            float* __restrict__ rcnt, int n)
{
    int i = blockIdx.x * blockDim.x + threadIdx.x;
    if (i < n) rcnt[i] = 1.0f / fmaxf(cnt[i], 1.0f);
}

// ----------------------------- TopK pooling --------------------------------
__global__ void norm_kernel(const float* __restrict__ w, float* __restrict__ out)
{
    __shared__ float red[256];
    float v = w[threadIdx.x];
    red[threadIdx.x] = v * v;
    __syncthreads();
    for (int s = 128; s > 0; s >>= 1) {
        if (threadIdx.x < s) red[threadIdx.x] += red[threadIdx.x + s];
        __syncthreads();
    }
    if (threadIdx.x == 0) out[0] = sqrtf(red[0]);
}

__global__ __launch_bounds__(256) void score_kernel(
    const float* __restrict__ x, const float* __restrict__ w,
    const float* __restrict__ normp, int n, float* __restrict__ score)
{
    int wave = threadIdx.x >> 5;
    int lane = threadIdx.x & 31;
    int i = blockIdx.x * 8 + wave;
    if (i >= n) return;
    const float* xr = x + (long long)i * 256;
    float s = 0.0f;
    #pragma unroll
    for (int t = 0; t < 8; ++t) s += xr[lane + 32 * t] * w[lane + 32 * t];
    #pragma unroll
    for (int off = 16; off > 0; off >>= 1) s += __shfl_xor(s, off, 32);
    if (lane == 0) score[i] = tanhf(s / normp[0]);
}

// stable descending rank: rank[i] = #{j : key_j > key_i || (== && j < i)}
__global__ __launch_bounds__(256) void rank_kernel(
    const float* __restrict__ score, unsigned* __restrict__ rank, int n)
{
    __shared__ unsigned skey[256];
    int i = blockIdx.x * 256 + threadIdx.x;
    unsigned ki = 0;
    if (i < n) {
        unsigned u = __float_as_uint(score[i]);
        ki = (u & 0x80000000u) ? ~u : (u | 0x80000000u);
    }
    unsigned r = 0;
    for (int base = 0; base < n; base += 256) {
        int j = base + threadIdx.x;
        unsigned kj = 0;
        if (j < n) {
            unsigned u = __float_as_uint(score[j]);
            kj = (u & 0x80000000u) ? ~u : (u | 0x80000000u);
        }
        skey[threadIdx.x] = kj;
        __syncthreads();
        int lim = (n - base < 256) ? (n - base) : 256;
        for (int t = 0; t < lim; ++t) {
            unsigned kt = skey[t];
            r += (unsigned)((kt > ki) || (kt == ki && (base + t) < i));
        }
        __syncthreads();
    }
    if (i < n) rank[i] = r;
}

__global__ void select_kernel(const float* __restrict__ score,
                              const unsigned* __restrict__ rank, int n, int k,
                              int* __restrict__ perm, float* __restrict__ topv,
                              int* __restrict__ keep, int* __restrict__ nmap)
{
    int i = blockIdx.x * blockDim.x + threadIdx.x;
    if (i >= n) return;
    unsigned r = rank[i];
    if (r < (unsigned)k) {
        perm[r] = i; topv[r] = score[i];
        keep[i] = 1; nmap[i] = (int)r;
    } else {
        keep[i] = 0; nmap[i] = 0;
    }
}

__global__ void pool_x_kernel(const float* __restrict__ x2,
                              const int* __restrict__ perm,
                              const float* __restrict__ topv, int k,
                              float* __restrict__ out)
{
    long long t = (long long)blockIdx.x * blockDim.x + threadIdx.x;
    if (t >= (long long)k * 64) return;
    int r = (int)(t >> 6);
    int f = (int)(t & 63) * 4;
    float tv = topv[r];
    float4 v = *(const float4*)(x2 + (long long)perm[r] * 256 + f);
    v.x *= tv; v.y *= tv; v.z *= tv; v.w *= tv;
    *(float4*)(out + (long long)r * 256 + f) = v;
}

__global__ void edge_update_kernel(const int* __restrict__ srcI,
                                   const int* __restrict__ dstI,
                                   const float* __restrict__ maskI,
                                   const int* __restrict__ keep,
                                   const int* __restrict__ nmap, int E,
                                   int* __restrict__ srcO, int* __restrict__ dstO,
                                   float* __restrict__ maskO)
{
    int e = blockIdx.x * blockDim.x + threadIdx.x;
    if (e >= E) return;
    int s = srcI[e], d = dstI[e];
    float mval = maskI ? maskI[e] : 1.0f;
    maskO[e] = mval * (float)(keep[s] & keep[d]);
    srcO[e] = nmap[s];
    dstO[e] = nmap[d];
}

// ----------------------------- unpool / concat -----------------------------
__global__ void unpool_kernel(const float* __restrict__ xv,
                              const int* __restrict__ perm, int k,
                              float* __restrict__ out)   // out pre-zeroed
{
    long long t = (long long)blockIdx.x * blockDim.x + threadIdx.x;
    if (t >= (long long)k * 64) return;
    int r = (int)(t >> 6);
    int f = (int)(t & 63) * 4;
    *(float4*)(out + (long long)perm[r] * 256 + f) =
        *(const float4*)(xv + (long long)r * 256 + f);
}

__global__ void concat_kernel(const float* __restrict__ a,
                              const float* __restrict__ b, int n,
                              float* __restrict__ out)   // out[n,512]=[a|b]
{
    long long t = (long long)blockIdx.x * blockDim.x + threadIdx.x;
    if (t >= (long long)n * 64) return;
    int r = (int)(t >> 6);
    int f = (int)(t & 63) * 4;
    *(float4*)(out + (long long)r * 512 + f) =
        *(const float4*)(a + (long long)r * 256 + f);
    *(float4*)(out + (long long)r * 512 + 256 + f) =
        *(const float4*)(b + (long long)r * 256 + f);
}

// ===========================================================================
extern "C" void kernel_launch(void* const* d_in, const int* in_sizes, int n_in,
                              void* d_out, int out_size, void* d_ws, size_t ws_size,
                              hipStream_t stream)
{
    (void)in_sizes; (void)n_in; (void)out_size; (void)ws_size;
    const float* x0   = (const float*)d_in[0];
    const int*   ei   = (const int*)d_in[1];
    const int*   src0 = ei;
    const int*   dst0 = ei + K_E;
    const float* pw   = (const float*)d_in[3];
    const float* dWl  = (const float*)d_in[4];
    const float* dWr  = (const float*)d_in[5];
    const float* db   = (const float*)d_in[6];
    const float* uWl1 = (const float*)d_in[7];
    const float* uWr1 = (const float*)d_in[8];
    const float* ub1  = (const float*)d_in[9];
    const float* uWl2 = (const float*)d_in[10];
    const float* uWr2 = (const float*)d_in[11];
    const float* ub2  = (const float*)d_in[12];
    const float* lw   = (const float*)d_in[13];
    const float* lb   = (const float*)d_in[14];
    float* dout = (float*)d_out;

    // ---- deterministic workspace layout ----
    char* base = (char*)d_ws;
    size_t off = 0;
    auto alloc = [&](size_t bytes) -> void* {
        off = (off + 255) & ~(size_t)255;
        void* p = base + off;
        off += bytes;
        return p;
    };
    float* AGG  = (float*)alloc((size_t)K_N0 * 512 * 4);
    float* HBUF = (float*)alloc((size_t)K_N0 * 512 * 4);
    float* XBUF = (float*)alloc((size_t)K_N0 * 256 * 4);
    float* TBUF = (float*)alloc((size_t)K_N0 * 256 * 4);
    float* XS1  = (float*)alloc((size_t)K_NS1 * 256 * 4);
    float* XS2  = (float*)alloc((size_t)K_NS2 * 256 * 4);
    float* X3   = (float*)alloc((size_t)K_NS3 * 256 * 4);
    float* CNT  = (float*)alloc((size_t)K_N0 * 4);
    float* RCNT = (float*)alloc((size_t)K_N0 * 4);
    float* SC   = (float*)alloc((size_t)K_N0 * 4);
    unsigned* RANK = (unsigned*)alloc((size_t)K_N0 * 4);
    int*   KEEP = (int*)alloc((size_t)K_N0 * 4);
    int*   NMAP = (int*)alloc((size_t)K_N0 * 4);
    int*   PERM0 = (int*)alloc((size_t)K_NS1 * 4);
    float* TOPV0 = (float*)alloc((size_t)K_NS1 * 4);
    int*   PERM1 = (int*)alloc((size_t)K_NS2 * 4);
    float* TOPV1 = (float*)alloc((size_t)K_NS2 * 4);
    int*   PERM2 = (int*)alloc((size_t)K_NS3 * 4);
    float* TOPV2 = (float*)alloc((size_t)K_NS3 * 4);
    int*   SRC1 = (int*)alloc((size_t)K_E * 4);
    int*   DST1 = (int*)alloc((size_t)K_E * 4);
    float* MSK1 = (float*)alloc((size_t)K_E * 4);
    int*   SRC2 = (int*)alloc((size_t)K_E * 4);
    int*   DST2 = (int*)alloc((size_t)K_E * 4);
    float* MSK2 = (float*)alloc((size_t)K_E * 4);
    float* NORM = (float*)alloc(256);

    const int HH = K_H * K_H;            // 65536
    const int H2 = K_H * 2 * K_H;        // 131072

    // bf16 weight shadows
    unsigned short* DWL_B  = (unsigned short*)alloc((size_t)6 * HH * 2);
    unsigned short* DWR_B  = (unsigned short*)alloc((size_t)6 * HH * 2);
    unsigned short* UWL1_B = (unsigned short*)alloc((size_t)2 * H2 * 2);
    unsigned short* UWR1_B = (unsigned short*)alloc((size_t)2 * H2 * 2);
    unsigned short* UWL2_B = (unsigned short*)alloc((size_t)2 * HH * 2);
    unsigned short* UWR2_B = (unsigned short*)alloc((size_t)2 * HH * 2);
    unsigned short* LW_B   = (unsigned short*)alloc((size_t)K_H * 1024 * 2);

    auto cdiv = [](long long a, long long b) { return (unsigned)((a + b - 1) / b); };

    auto cvtw = [&](const float* in, unsigned short* out, long long nElems) {
        int nPairs = (int)(nElems >> 1);
        cvt_bf16_kernel<<<cdiv(nPairs, 256), 256, 0, stream>>>(in, (unsigned*)out, nPairs);
    };
    cvtw(dWl,  DWL_B,  (long long)6 * HH);
    cvtw(dWr,  DWR_B,  (long long)6 * HH);
    cvtw(uWl1, UWL1_B, (long long)2 * H2);
    cvtw(uWr1, UWR1_B, (long long)2 * H2);
    cvtw(uWl2, UWL2_B, (long long)2 * HH);
    cvtw(uWr2, UWR2_B, (long long)2 * HH);
    cvtw(lw,   LW_B,   (long long)K_H * 1024);

    auto aggregate = [&](const float* x, const int* src, const int* dst,
                         const float* mask, int n, int F) {
        hipMemsetAsync(AGG, 0, (size_t)n * F * 4, stream);
        hipMemsetAsync(CNT, 0, (size_t)n * 4, stream);
        long long tot = (long long)K_E * (F >> 2);
        agg_feat_kernel<<<cdiv(tot, 256), 256, 0, stream>>>(x, src, dst, mask, AGG, K_E, F);
        agg_cnt_kernel<<<cdiv(K_E, 256), 256, 0, stream>>>(dst, mask, CNT, K_E);
        rcnt_kernel<<<cdiv(n, 256), 256, 0, stream>>>(CNT, RCNT, n);
    };
    auto gemm = [&](const float* A1, const float* rs1,
                    const unsigned short* W1, int K1, int ldw1,
                    const float* A2, const unsigned short* W2, int K2, int ldw2,
                    const float* bias, float* Cp, int M, int relu, int addC) {
        dim3 g(cdiv(M, 64), 2);
        wmma_gemm_kernel<<<g, 256, 0, stream>>>(A1, rs1, W1, K1, ldw1,
                                                A2, W2, K2, ldw2, bias, Cp, M, relu, addC);
    };
    auto sage = [&](const float* xin, const int* src, const int* dst, const float* mask,
                    int n, int K, const unsigned short* Wl, const unsigned short* Wr,
                    const float* b, float* outp, int relu) {
        aggregate(xin, src, dst, mask, n, K);
        gemm(AGG, RCNT, Wl, K, K, xin, Wr, K, K, b, outp, n, relu, 0);
    };
    auto topk = [&](const float* x2, const float* w, int n, int k,
                    int* perm, float* topv) {
        norm_kernel<<<1, 256, 0, stream>>>(w, NORM);
        score_kernel<<<cdiv(n, 8), 256, 0, stream>>>(x2, w, NORM, n, SC);
        rank_kernel<<<cdiv(n, 256), 256, 0, stream>>>(SC, RANK, n);
        select_kernel<<<cdiv(n, 256), 256, 0, stream>>>(SC, RANK, n, k, perm, topv, KEEP, NMAP);
    };

    // ================= encoder level 0 (n=50000) =================
    sage(x0,   src0, dst0, 0, K_N0, 256, DWL_B + 0 * HH, DWR_B + 0 * HH, db + 0 * K_H, HBUF, 1);
    sage(HBUF, src0, dst0, 0, K_N0, 256, DWL_B + 1 * HH, DWR_B + 1 * HH, db + 1 * K_H, XBUF, 0);
    topk(XBUF, pw + 0 * K_H, K_N0, K_NS1, PERM0, TOPV0);
    pool_x_kernel<<<cdiv((long long)K_NS1 * 64, 256), 256, 0, stream>>>(XBUF, PERM0, TOPV0, K_NS1, XS1);
    edge_update_kernel<<<cdiv(K_E, 256), 256, 0, stream>>>(src0, dst0, 0, KEEP, NMAP, K_E, SRC1, DST1, MSK1);

    // ================= encoder level 1 (n=25000) =================
    sage(XS1,  SRC1, DST1, MSK1, K_NS1, 256, DWL_B + 2 * HH, DWR_B + 2 * HH, db + 2 * K_H, HBUF, 1);
    sage(HBUF, SRC1, DST1, MSK1, K_NS1, 256, DWL_B + 3 * HH, DWR_B + 3 * HH, db + 3 * K_H, XBUF, 0);
    topk(XBUF, pw + 1 * K_H, K_NS1, K_NS2, PERM1, TOPV1);
    pool_x_kernel<<<cdiv((long long)K_NS2 * 64, 256), 256, 0, stream>>>(XBUF, PERM1, TOPV1, K_NS2, XS2);
    edge_update_kernel<<<cdiv(K_E, 256), 256, 0, stream>>>(SRC1, DST1, MSK1, KEEP, NMAP, K_E, SRC2, DST2, MSK2);

    // ================= encoder level 2 (n=12500) =================
    sage(XS2,  SRC2, DST2, MSK2, K_NS2, 256, DWL_B + 4 * HH, DWR_B + 4 * HH, db + 4 * K_H, HBUF, 1);
    sage(HBUF, SRC2, DST2, MSK2, K_NS2, 256, DWL_B + 5 * HH, DWR_B + 5 * HH, db + 5 * K_H, XBUF, 0);
    topk(XBUF, pw + 2 * K_H, K_NS2, K_NS3, PERM2, TOPV2);
    pool_x_kernel<<<cdiv((long long)K_NS3 * 64, 256), 256, 0, stream>>>(XBUF, PERM2, TOPV2, K_NS3, X3);

    // ================= decoder i=1 (n=25000) =================
    hipMemsetAsync(TBUF, 0, (size_t)K_NS2 * 256 * 4, stream);
    unpool_kernel<<<cdiv((long long)K_NS3 * 64, 256), 256, 0, stream>>>(X3, PERM2, K_NS3, TBUF);
    hipMemsetAsync(XBUF, 0, (size_t)K_NS1 * 256 * 4, stream);
    unpool_kernel<<<cdiv((long long)K_NS2 * 64, 256), 256, 0, stream>>>(TBUF, PERM1, K_NS2, XBUF);
    concat_kernel<<<cdiv((long long)K_NS1 * 64, 256), 256, 0, stream>>>(XBUF, XS1, K_NS1, HBUF);
    aggregate(HBUF, SRC1, DST1, MSK1, K_NS1, 512);
    gemm(AGG, RCNT, UWL1_B + 1 * H2, 512, 512, HBUF, UWR1_B + 1 * H2, 512, 512,
         ub1 + 1 * K_H, TBUF, K_NS1, 1, 0);
    sage(TBUF, SRC1, DST1, MSK1, K_NS1, 256, UWL2_B + 1 * HH, UWR2_B + 1 * HH, ub2 + 1 * K_H, XBUF, 0);

    // ================= decoder i=0 (n=50000) =================
    hipMemsetAsync(TBUF, 0, (size_t)K_N0 * 256 * 4, stream);
    unpool_kernel<<<cdiv((long long)K_NS1 * 64, 256), 256, 0, stream>>>(XBUF, PERM0, K_NS1, TBUF);
    concat_kernel<<<cdiv((long long)K_N0 * 64, 256), 256, 0, stream>>>(TBUF, x0, K_N0, HBUF);
    aggregate(HBUF, src0, dst0, 0, K_N0, 512);
    gemm(AGG, RCNT, UWL1_B + 0, 512, 512, HBUF, UWR1_B + 0, 512, 512,
         ub1 + 0, TBUF, K_N0, 1, 0);
    sage(TBUF, src0, dst0, 0, K_N0, 256, UWL2_B + 0, UWR2_B + 0, ub2 + 0, XBUF, 0);
    // XBUF = final decoder features at level 0

    // ================= JumpingKnowledge 'cat' + lin1 =================
    // up1 = unpool(XS1, perm0) -> AGG ; up2 = unpool(unpool(XS2,perm1),perm0) -> HBUF
    hipMemsetAsync(AGG, 0, (size_t)K_N0 * 256 * 4, stream);
    unpool_kernel<<<cdiv((long long)K_NS1 * 64, 256), 256, 0, stream>>>(XS1, PERM0, K_NS1, AGG);
    hipMemsetAsync(TBUF, 0, (size_t)K_NS1 * 256 * 4, stream);
    unpool_kernel<<<cdiv((long long)K_NS2 * 64, 256), 256, 0, stream>>>(XS2, PERM1, K_NS2, TBUF);
    hipMemsetAsync(HBUF, 0, (size_t)K_N0 * 256 * 4, stream);
    unpool_kernel<<<cdiv((long long)K_NS1 * 64, 256), 256, 0, stream>>>(TBUF, PERM0, K_NS1, HBUF);

    // out = relu(x0@W0^T + up1@W1^T + up2@W2^T + xfinal@W3^T + b)
    // lin1_w is (256, 1024): block k lives at columns k*256.. -> fold into base ptr
    gemm(x0,   0, LW_B + 0,   256, 1024, AGG,  LW_B + 256, 256, 1024, 0,  dout, K_N0, 0, 0);
    gemm(HBUF, 0, LW_B + 512, 256, 1024, XBUF, LW_B + 768, 256, 1024, lb, dout, K_N0, 1, 1);
}
